// CrossModalContextMultiHeadedAttention_88502096101977
// MI455X (gfx1250) — matrix operand order repaired and checked by reading, not compiled, not run on hardware
//
#include <hip/hip_runtime.h>
#include <hip/hip_bf16.h>
#include <stdint.h>

// ---------------------------------------------------------------------------
// CrossModalContextMultiHeadedAttention for MI455X (gfx1250, wave32, WMMA)
//
// B=4, N=1024, H=8, D=128.
// bf16 WMMA (v_wmma_f32_16x16x32_bf16) for every GEMM, f32 softmax.
// Outputs dominate traffic (~151MB) -> memory-bound; bf16 K=32 WMMA keeps the
// ~31 GFLOP of matmul under the ~11us HBM floor (23.3 TB/s).
// All fragment loads are aligned 16B (b128) accesses:
//   - weights pre-converted to bf16 and TRANSPOSED once (contiguous B-frags)
//   - vv additionally stored transposed [B,H,D,N] for contiguous PV B-frags
// text_value projection is dead code in the reference -> skipped.
// ---------------------------------------------------------------------------

#define B_ 4
#define N_ 1024
#define H_ 8
#define D_ 128
#define SCALE_ 0.088388347648318447f  // 1/sqrt(128)

typedef __attribute__((ext_vector_type(16))) __bf16 v16bf;
typedef __attribute__((ext_vector_type(8)))  float  v8f;

union AFrag {                 // 16 bf16 per lane = 8 VGPRs = 2 x b128
  v16bf  v;
  __bf16 b[16];
  uint4  q[2];
};
union Acc { v8f v; float f[8]; };   // 16x16 f32 C/D tile = 8 VGPRs

__device__ __forceinline__ v8f wmma_bf16(v16bf a, v16bf b, v8f c) {
  // 8 args: (neg_a, A, neg_b, B, c_mod, C, reuse_a, reuse_b)
  return __builtin_amdgcn_wmma_f32_16x16x32_bf16(false, a, false, b,
                                                 (short)0, c, false, false);
}

// ---------------------------------------------------------------------------
// One-time weight prep: bf16 + transpose, so projection B-fragments are
// contiguous per lane.  WT[e*D + d] = bf16(W[d*D + e]).
// ---------------------------------------------------------------------------
__global__ __launch_bounds__(256)
void weights_prep_kernel(const float* __restrict__ vW, const float* __restrict__ tW,
                         __bf16* __restrict__ WvT, __bf16* __restrict__ WtT) {
  const int i = blockIdx.x * 256 + threadIdx.x;   // 0 .. 16383
  const int d = i >> 7, e = i & 127;
  WvT[e * D_ + d] = (__bf16)vW[d * D_ + e];
  WtT[e * D_ + d] = (__bf16)tW[d * D_ + e];
}

// ---------------------------------------------------------------------------
// Projection: Y[b,h,n,:] = X[b,n,h,:] @ W + bias, bf16, layout [B,H,N,D].
// Optional second output YT (for vv): transposed layout [B,H,D,N].
// Block: 16 rows of flat [B*N*H]; wave w handles output cols [16w,16w+16).
// A-frag (16-bit A 16x32): lanes 0-15 M=lane, K kb..kb+15;
// lanes 16-31: K kb+16..kb+31.  B-frag (32x16): N=lane%16,
// K = kb + 16*(lane/16) + j  (contiguous -> 2 x b128 from WT row).
// ---------------------------------------------------------------------------
__global__ __launch_bounds__(256)
void proj_bf16_kernel(const float* __restrict__ X, const __bf16* __restrict__ WT,
                      const float* __restrict__ bias, __bf16* __restrict__ Y,
                      __bf16* __restrict__ YT) {
  const int lane = threadIdx.x & 31;
  const int wave = threadIdx.x >> 5;
  const int half = lane >> 4;
  const int l16  = lane & 15;
  const int m0 = blockIdx.x * 16;
  const int e0 = wave * 16;

  const float*  xrow = X + (size_t)(m0 + l16) * D_;
  const __bf16* wrow = WT + (size_t)(e0 + l16) * D_;

  Acc c;
#pragma unroll
  for (int i = 0; i < 8; ++i) c.f[i] = 0.f;

#pragma unroll
  for (int ks = 0; ks < 4; ++ks) {
    const int kb = ks * 32;
    AFrag a, bm;
    // A: 16 f32 -> bf16 (two 8-float groups, each 2 x float4 loads)
    const float4* xa = (const float4*)(xrow + kb + half * 16);
    const float4* xb = (const float4*)(xrow + kb + half * 16 + 8);
    float4 f0 = xa[0], f1 = xa[1], f2 = xb[0], f3 = xb[1];
    a.b[0]  = (__bf16)f0.x;  a.b[1]  = (__bf16)f0.y;
    a.b[2]  = (__bf16)f0.z;  a.b[3]  = (__bf16)f0.w;
    a.b[4]  = (__bf16)f1.x;  a.b[5]  = (__bf16)f1.y;
    a.b[6]  = (__bf16)f1.z;  a.b[7]  = (__bf16)f1.w;
    a.b[8]  = (__bf16)f2.x;  a.b[9]  = (__bf16)f2.y;
    a.b[10] = (__bf16)f2.z;  a.b[11] = (__bf16)f2.w;
    a.b[12] = (__bf16)f3.x;  a.b[13] = (__bf16)f3.y;
    a.b[14] = (__bf16)f3.z;  a.b[15] = (__bf16)f3.w;
    // B: 16 contiguous bf16 from transposed weights
    const uint4* wq = (const uint4*)(wrow + kb + half * 16);
    bm.q[0] = wq[0];
    bm.q[1] = wq[1];
    c.v = wmma_bf16(a.v, bm.v, c.v);
  }

  const float be = bias[e0 + l16];
#pragma unroll
  for (int r = 0; r < 8; ++r) {
    const int m   = m0 + r + half * 8;        // C/D: lanes16-31 hold M=r+8
    const int b   = m / (N_ * H_);
    const int rem = m % (N_ * H_);
    const int n   = rem / H_;
    const int h   = rem % H_;
    const __bf16 val = (__bf16)(c.f[r] + be);
    Y[((size_t)(b * H_ + h) * N_ + n) * D_ + e0 + l16] = val;
    if (YT != nullptr)                        // transposed copy (vv only)
      YT[((size_t)(b * H_ + h) * D_ + e0 + l16) * N_ + n] = val;
  }
}

// ---------------------------------------------------------------------------
// t2v attention fused with @vv.  One block per (b, h, 16 text rows).
// Stage 1: S[16,1024] bf16 scores in LDS (each wave: 8 key tiles x 4 WMMA).
// Stage 2: row softmax in f32; P rewritten in place (bf16).
// Stage 3: O = P @ Vv via vvT (contiguous b128 B-frags); normalize; store f32.
// ---------------------------------------------------------------------------
__global__ __launch_bounds__(256)
void attn_t2v_kernel(const __bf16* __restrict__ tqP,
                     const __bf16* __restrict__ vkP,
                     const __bf16* __restrict__ vvT,
                     float* __restrict__ out) {
  __shared__ __bf16 S[16 * N_];     // 32 KB: scores, then probabilities
  __shared__ float red[16 * 16];
  __shared__ float rowmax[16];
  __shared__ float rowsum[16];

  const int lane = threadIdx.x & 31, wave = threadIdx.x >> 5;
  const int half = lane >> 4, l16 = lane & 15;
  const int x = blockIdx.x;
  const int tile = x & 63, h = (x >> 6) & 7, b = x >> 9;
  const int t0 = tile * 16;
  const __bf16* tq = tqP + (size_t)(b * H_ + h) * N_ * D_;
  const __bf16* vk = vkP + (size_t)(b * H_ + h) * N_ * D_;
  const __bf16* vt = vvT + (size_t)(b * H_ + h) * D_ * N_;

  // Query A-fragments for this 16-row tile (reused across all 64 key tiles).
  AFrag aq[4];
  {
    const __bf16* trow = tq + (size_t)(t0 + l16) * D_;
#pragma unroll
    for (int ks = 0; ks < 4; ++ks) {
      aq[ks].q[0] = *(const uint4*)(trow + ks * 32 + half * 16);
      aq[ks].q[1] = *(const uint4*)(trow + ks * 32 + half * 16 + 8);
    }
  }

  // ---- Stage 1: scores -> LDS ----
  for (int jt = 0; jt < 8; ++jt) {
    const int v0 = (wave * 8 + jt) * 16;
    const __bf16* vrow = vk + (size_t)(v0 + l16) * D_;
    Acc c;
#pragma unroll
    for (int i = 0; i < 8; ++i) c.f[i] = 0.f;
#pragma unroll
    for (int ks = 0; ks < 4; ++ks) {
      AFrag bm;  // B[d][v]=vk[v][d]: 16 contiguous d per lane
      bm.q[0] = *(const uint4*)(vrow + ks * 32 + half * 16);
      bm.q[1] = *(const uint4*)(vrow + ks * 32 + half * 16 + 8);
      c.v = wmma_bf16(aq[ks].v, bm.v, c.v);
    }
#pragma unroll
    for (int r = 0; r < 8; ++r)
      S[(r + half * 8) * N_ + v0 + l16] = (__bf16)(c.f[r] * SCALE_);
  }
  __syncthreads();

  // ---- Stage 2: softmax stats (thread = (row, 64-elem segment)) ----
  const int row = threadIdx.x >> 4, seg = threadIdx.x & 15;
  const int base = row * N_ + seg * 64;
  float mx = -3.4e38f;
  for (int i = 0; i < 64; ++i) mx = fmaxf(mx, (float)S[base + i]);
  red[row * 16 + seg] = mx;
  __syncthreads();
  if (seg == 0) {
    float m2 = red[row * 16];
    for (int i = 1; i < 16; ++i) m2 = fmaxf(m2, red[row * 16 + i]);
    rowmax[row] = m2;
  }
  __syncthreads();
  const float rm = rowmax[row];
  float sum = 0.f;
  for (int i = 0; i < 64; ++i) {
    const float e = __expf((float)S[base + i] - rm);
    sum += e;
    S[base + i] = (__bf16)e;   // in place: each element owned by one thread
  }
  red[row * 16 + seg] = sum;
  __syncthreads();
  if (seg == 0) {
    float s2 = 0.f;
    for (int i = 0; i < 16; ++i) s2 += red[row * 16 + i];
    rowsum[row] = s2;
  }
  __syncthreads();

  // ---- Stage 3: O = P @ Vv ----
  const int d0 = wave * 16;
  Acc o;
#pragma unroll
  for (int i = 0; i < 8; ++i) o.f[i] = 0.f;
  const __bf16* vcol = vt + (size_t)(d0 + l16) * N_;   // vvT row = output col
  for (int kv = 0; kv < 32; ++kv) {
    const int kb = kv * 32;
    AFrag pa, bv;
    // P from LDS: row = l16, two aligned b128 reads
    pa.q[0] = *(const uint4*)(S + l16 * N_ + kb + half * 16);
    pa.q[1] = *(const uint4*)(S + l16 * N_ + kb + half * 16 + 8);
    // V: B[v][d] = vvT[d][v], 16 contiguous v per lane
    bv.q[0] = *(const uint4*)(vcol + kb + half * 16);
    bv.q[1] = *(const uint4*)(vcol + kb + half * 16 + 8);
    if (kv + 1 < 32)                  // uniform; emits global_prefetch_b8
      __builtin_prefetch(vcol + kb + 32 + half * 16, 0, 1);
    o.v = wmma_bf16(pa.v, bv.v, o.v);
  }
#pragma unroll
  for (int r = 0; r < 8; ++r) {
    const int tl = r + half * 8;
    const float val = o.f[r] / rowsum[tl];
    out[((size_t)(b * N_ + t0 + tl) * H_ + h) * D_ + d0 + l16] = val;
  }
}

// ---------------------------------------------------------------------------
// v2t attention scores (full softmaxed matrix is an output).
// One block per (b, h, 16 visual-query rows); cols = 1024 text keys.
// ---------------------------------------------------------------------------
__global__ __launch_bounds__(256)
void attn_v2t_kernel(const __bf16* __restrict__ vqP,
                     const __bf16* __restrict__ tkP,
                     float* __restrict__ out) {
  __shared__ __bf16 S[16 * N_];
  __shared__ float red[16 * 16];
  __shared__ float rowmax[16];
  __shared__ float rowsum[16];

  const int lane = threadIdx.x & 31, wave = threadIdx.x >> 5;
  const int half = lane >> 4, l16 = lane & 15;
  const int x = blockIdx.x;
  const int tile = x & 63, h = (x >> 6) & 7, b = x >> 9;
  const int v0r = tile * 16;
  const __bf16* vq = vqP + (size_t)(b * H_ + h) * N_ * D_;
  const __bf16* tk = tkP + (size_t)(b * H_ + h) * N_ * D_;

  AFrag aq[4];
  {
    const __bf16* vrow = vq + (size_t)(v0r + l16) * D_;
#pragma unroll
    for (int ks = 0; ks < 4; ++ks) {
      aq[ks].q[0] = *(const uint4*)(vrow + ks * 32 + half * 16);
      aq[ks].q[1] = *(const uint4*)(vrow + ks * 32 + half * 16 + 8);
    }
  }

  for (int jt = 0; jt < 8; ++jt) {
    const int c0 = (wave * 8 + jt) * 16;
    const __bf16* krow = tk + (size_t)(c0 + l16) * D_;
    Acc c;
#pragma unroll
    for (int i = 0; i < 8; ++i) c.f[i] = 0.f;
#pragma unroll
    for (int ks = 0; ks < 4; ++ks) {
      AFrag bm;
      bm.q[0] = *(const uint4*)(krow + ks * 32 + half * 16);
      bm.q[1] = *(const uint4*)(krow + ks * 32 + half * 16 + 8);
      c.v = wmma_bf16(aq[ks].v, bm.v, c.v);
    }
#pragma unroll
    for (int r = 0; r < 8; ++r)
      S[(r + half * 8) * N_ + c0 + l16] = (__bf16)(c.f[r] * SCALE_);
  }
  __syncthreads();

  const int row = threadIdx.x >> 4, seg = threadIdx.x & 15;
  const int base = row * N_ + seg * 64;
  float mx = -3.4e38f;
  for (int i = 0; i < 64; ++i) mx = fmaxf(mx, (float)S[base + i]);
  red[row * 16 + seg] = mx;
  __syncthreads();
  if (seg == 0) {
    float m2 = red[row * 16];
    for (int i = 1; i < 16; ++i) m2 = fmaxf(m2, red[row * 16 + i]);
    rowmax[row] = m2;
  }
  __syncthreads();
  const float rm = rowmax[row];
  float sum = 0.f;
  for (int i = 0; i < 64; ++i) {
    const float e = __expf((float)S[base + i] - rm);
    sum += e;
    S[base + i] = (__bf16)e;
  }
  red[row * 16 + seg] = sum;
  __syncthreads();
  if (seg == 0) {
    float s2 = 0.f;
    for (int i = 0; i < 16; ++i) s2 += red[row * 16 + i];
    rowsum[row] = s2;
  }
  __syncthreads();

  const float inv = 1.f / rowsum[row];
  float* orow = out + ((size_t)(b * H_ + h) * N_ + (v0r + row)) * (size_t)N_
                    + seg * 64;
#pragma unroll 4
  for (int i = 0; i < 16; ++i) {    // 16 x float4 = 64 f32, b128 stores
    float4 v;
    v.x = (float)S[base + 4 * i + 0] * inv;
    v.y = (float)S[base + 4 * i + 1] * inv;
    v.z = (float)S[base + 4 * i + 2] * inv;
    v.w = (float)S[base + 4 * i + 3] * inv;
    ((float4*)orow)[i] = v;
  }
}

// ---------------------------------------------------------------------------
extern "C" void kernel_launch(void* const* d_in, const int* in_sizes, int n_in,
                              void* d_out, int out_size, void* d_ws, size_t ws_size,
                              hipStream_t stream) {
  (void)in_sizes; (void)n_in; (void)out_size; (void)ws_size;

  const float* vq = (const float*)d_in[0];
  const float* vk = (const float*)d_in[1];
  const float* vv = (const float*)d_in[2];
  const float* tq = (const float*)d_in[3];
  const float* tk = (const float*)d_in[4];
  // d_in[5] (text_value): projection result never reaches an output -> skipped
  const float* vW = (const float*)d_in[6];
  const float* vb = (const float*)d_in[7];
  const float* tW = (const float*)d_in[8];
  const float* tb = (const float*)d_in[9];

  const size_t SZ = (size_t)B_ * H_ * N_ * D_;   // 4,194,304 elems
  __bf16* ws  = (__bf16*)d_ws;                   // ~50 MB of bf16 scratch
  __bf16* vqP = ws + 0 * SZ;
  __bf16* vkP = ws + 1 * SZ;
  __bf16* vvP = ws + 2 * SZ;
  __bf16* tqP = ws + 3 * SZ;
  __bf16* tkP = ws + 4 * SZ;
  __bf16* vvT = ws + 5 * SZ;                     // [B,H,D,N]
  __bf16* WvT = ws + 6 * SZ;                     // [D,D] transposed bf16
  __bf16* WtT = WvT + (size_t)D_ * D_;

  const dim3 blk(256);
  weights_prep_kernel<<<dim3(D_ * D_ / 256), blk, 0, stream>>>(vW, tW, WvT, WtT);

  const dim3 pgrid(B_ * N_ * H_ / 16);           // 2048 row tiles
  proj_bf16_kernel<<<pgrid, blk, 0, stream>>>(vq, WvT, vb, vqP, nullptr);
  proj_bf16_kernel<<<pgrid, blk, 0, stream>>>(vk, WvT, vb, vkP, nullptr);
  proj_bf16_kernel<<<pgrid, blk, 0, stream>>>(vv, WvT, vb, vvP, vvT);
  proj_bf16_kernel<<<pgrid, blk, 0, stream>>>(tq, WtT, tb, tqP, nullptr);
  proj_bf16_kernel<<<pgrid, blk, 0, stream>>>(tk, WtT, tb, tkP, nullptr);

  const dim3 agrid(B_ * H_ * (N_ / 16));         // 2048 (b,h,row-tile) blocks
  attn_t2v_kernel<<<agrid, blk, 0, stream>>>(tqP, vkP, vvT, (float*)d_out);
  attn_v2t_kernel<<<agrid, blk, 0, stream>>>(vqP, tkP, (float*)d_out + SZ);
}
